// RNN_14310831030903
// MI455X (gfx1250) — compile-verified, matrix operand order checked by reading, not run on hardware
//
#include <hip/hip_runtime.h>
#include <hip/hip_bf16.h>
#include <math.h>
#include <stdint.h>

#define VOCAB 32000
#define EMB   1024
#define BB    8
#define TT    512
#define ROWS  (BB*TT)   // 4096

#define KCH        32   // K-chunk per WMMA step
#define LDS_STRIDE 40   // 32 bf16 + 8 pad: 80B rows -> 16B aligned, bank-conflict-free

typedef __attribute__((ext_vector_type(16))) __bf16 v16bf;
typedef __attribute__((ext_vector_type(8)))  float  v8f;
typedef __attribute__((ext_vector_type(4)))  float  v4f;
typedef __attribute__((ext_vector_type(4)))  unsigned int v4u;

union FragBF {
    v16bf v;
    v4u   q[2];
    unsigned short s[16];
};

static __device__ __forceinline__ unsigned short f32_to_bf16(float f) {
    union { float f; unsigned int u; } in;
    in.f = f;
    unsigned int u = in.u;
    u += ((u >> 16) & 1u) + 0x7FFFu;   // round to nearest even
    return (unsigned short)(u >> 16);
}

// --------------------------- async copy plumbing ---------------------------
#if __has_builtin(__builtin_amdgcn_global_load_async_to_lds_b128) && \
    __has_builtin(__builtin_amdgcn_s_wait_asynccnt)
#define USE_ASYNC_LDS 1
typedef int v4i __attribute__((vector_size(16)));
typedef __attribute__((address_space(1))) v4i* g_v4i_p;   // global src
typedef __attribute__((address_space(3))) v4i* l_v4i_p;   // LDS dst
static __device__ __forceinline__ void async_cp16(const unsigned short* g,
                                                  unsigned short* l) {
    __builtin_amdgcn_global_load_async_to_lds_b128((g_v4i_p)g, (l_v4i_p)l, 0, 0);
}
#else
#define USE_ASYNC_LDS 0
static __device__ __forceinline__ void async_cp16(const unsigned short* g,
                                                  unsigned short* l) {
    *(v4u*)l = *(const v4u*)g;   // sync fallback: global->VGPR->LDS
}
#endif

// ---------------------------------------------------------------------------
// Transpose + convert: W [K][N] f32  ->  Wt [N][K] bf16   (LDS tiled 32x32)
// ---------------------------------------------------------------------------
__global__ __launch_bounds__(256) void transpose_cvt(
    const float* __restrict__ W, unsigned short* __restrict__ Wt, int K, int N)
{
    __shared__ float tile[32][33];
    const int tx = threadIdx.x & 31;
    const int ty = threadIdx.x >> 5;          // 0..7
    const int nBase = blockIdx.x * 32;
    const int kBase = blockIdx.y * 32;
#pragma unroll
    for (int i = 0; i < 4; ++i) {
        int k = kBase + ty + i * 8;
        tile[ty + i * 8][tx] = W[(long)k * N + (nBase + tx)];
    }
    __syncthreads();
#pragma unroll
    for (int i = 0; i < 4; ++i) {
        int n = nBase + ty + i * 8;
        Wt[(long)n * K + (kBase + tx)] = f32_to_bf16(tile[tx][ty + i * 8]);
    }
}

__global__ void zero_f32(float* __restrict__ p, int n) {
    int i = blockIdx.x * blockDim.x + threadIdx.x;
    if (i < n) p[i] = 0.0f;
}

// ---------------------------------------------------------------------------
// Input projection: xw[T,B,C] = emb[idx] @ W_xh + b_h     (WMMA bf16, f32 acc)
// Block tile 64x64, 8 waves, wave tile 16x32 (2 accumulators).
// ---------------------------------------------------------------------------
__global__ __launch_bounds__(256) void input_proj(
    const int* __restrict__ idx, const float* __restrict__ emb,
    const unsigned short* __restrict__ WxhT, const float* __restrict__ b_h,
    float* __restrict__ xw)
{
    const int lane  = threadIdx.x & 31;
    const int wave  = threadIdx.x >> 5;
    const int waveM = wave & 3;      // 0..3  (16 rows each)
    const int waveN = wave >> 2;     // 0..1  (32 cols each)
    const int g     = lane >> 4;     // half-wave select
    const int mr    = lane & 15;

    const int m = blockIdx.y * 64 + waveM * 16 + mr;   // row in [0,4096)
    const int t = m >> 3;
    const int b = m & 7;
    const int erow = idx[b * TT + t];                   // idx is [B,T]
    const float* arow = emb + (long)erow * EMB;

    const int n0 = blockIdx.x * 64 + waveN * 32 + mr;
    const int n1 = n0 + 16;
    const unsigned short* brow0 = WxhT + (long)n0 * EMB;
    const unsigned short* brow1 = WxhT + (long)n1 * EMB;

    v8f acc0 = {}; v8f acc1 = {};

    for (int kk = 0; kk < EMB; kk += KCH) {
        const int o0 = kk + 8 * g;
        const int o1 = kk + 16 + 8 * g;
        FragBF a;
        v4f f0 = *(const v4f*)(arow + o0);
        v4f f1 = *(const v4f*)(arow + o0 + 4);
        v4f f2 = *(const v4f*)(arow + o1);
        v4f f3 = *(const v4f*)(arow + o1 + 4);
#pragma unroll
        for (int j = 0; j < 4; ++j) {
            a.s[j]      = f32_to_bf16(f0[j]);
            a.s[4 + j]  = f32_to_bf16(f1[j]);
            a.s[8 + j]  = f32_to_bf16(f2[j]);
            a.s[12 + j] = f32_to_bf16(f3[j]);
        }
        FragBF fb0, fb1;
        fb0.q[0] = *(const v4u*)(brow0 + o0);
        fb0.q[1] = *(const v4u*)(brow0 + o1);
        fb1.q[0] = *(const v4u*)(brow1 + o0);
        fb1.q[1] = *(const v4u*)(brow1 + o1);

        acc0 = __builtin_amdgcn_wmma_f32_16x16x32_bf16(false, a.v, false, fb0.v,
                                                       (short)0, acc0, false, false);
        acc1 = __builtin_amdgcn_wmma_f32_16x16x32_bf16(false, a.v, false, fb1.v,
                                                       (short)0, acc1, false, false);
    }

    const int mBase = blockIdx.y * 64 + waveM * 16 + 8 * g;
    const float bh0 = b_h[n0];
    const float bh1 = b_h[n1];
#pragma unroll
    for (int r = 0; r < 8; ++r) {
        const int mo = mBase + r;
        xw[(long)mo * EMB + n0] = acc0[r] + bh0;
        xw[(long)mo * EMB + n1] = acc1[r] + bh1;
    }
}

// ---------------------------------------------------------------------------
// One recurrence step (fp32): h_next = tanh(xw[t] + h_prev @ W_hh)
// Also emits bf16 hs row for the output GEMM, laid out [B*T, C] (row b*T+t).
// ---------------------------------------------------------------------------
__global__ __launch_bounds__(256) void rnn_step(
    const float* __restrict__ xw, const float* __restrict__ Whh,
    const float* __restrict__ h_prev, float* __restrict__ h_next,
    unsigned short* __restrict__ hs_bf16, int t)
{
    const int j = blockIdx.x * blockDim.x + threadIdx.x; // 0..8191
    const int b = j >> 10;
    const int n = j & 1023;

    float acc = xw[(long)(t * BB + b) * EMB + n];
    const float* hp = h_prev + b * EMB;
#pragma unroll 8
    for (int k = 0; k < EMB; ++k)
        acc = fmaf(hp[k], Whh[(long)k * EMB + n], acc);

    const float h = tanhf(acc);
    h_next[b * EMB + n] = h;
    hs_bf16[(long)(b * TT + t) * EMB + n] = f32_to_bf16(h);
}

// ---------------------------------------------------------------------------
// Output GEMM: logits[B*T, V] = hs_bf16 @ W_out + b_out   (WMMA bf16, f32 acc)
// M = 4096, N = 32000, K = 1024.  Block tile 128x128, 8 waves (4M x 2N),
// wave tile 32x64 -> 8 accumulators, 8 wmma per K-chunk.
// A/B K-slices double-buffered in LDS via async global->LDS copies.
// ---------------------------------------------------------------------------
__global__ __launch_bounds__(256) void out_gemm(
    const unsigned short* __restrict__ A,   // hs_bf16 [4096][1024]
    const unsigned short* __restrict__ Bt,  // WoutT   [32000][1024]
    const float* __restrict__ b_out,
    float* __restrict__ logits)             // [4096][32000]
{
    __shared__ __align__(16) unsigned short As[2][128 * LDS_STRIDE];
    __shared__ __align__(16) unsigned short Bs[2][128 * LDS_STRIDE];

    const int tid   = threadIdx.x;
    const int lane  = tid & 31;
    const int wave  = tid >> 5;
    const int waveM = wave & 3;     // 4 M-waves * 32 rows
    const int waveN = wave >> 2;    // 2 N-waves * 64 cols
    const int g     = lane >> 4;
    const int mr    = lane & 15;

    const int mBase = blockIdx.y * 128 + waveM * 32;
    const int nBase = blockIdx.x * 128 + waveN * 64;

    // --- staging: each thread copies 4x16B per K-chunk (A:2, B:2) ---
    const int crow = tid >> 1;           // 0..127
    const int cseg = (tid & 1) * 8;      // 0 / 8
    const unsigned short* gA = A  + (long)(blockIdx.y * 128 + crow) * EMB;
    const unsigned short* gB = Bt + (long)(blockIdx.x * 128 + crow) * EMB;
    const int lbase = crow * LDS_STRIDE + cseg;

    auto stage = [&](int buf, int kk) {
        async_cp16(gA + kk + cseg,      &As[buf][lbase]);
        async_cp16(gA + kk + 16 + cseg, &As[buf][lbase + 16]);
        async_cp16(gB + kk + cseg,      &Bs[buf][lbase]);
        async_cp16(gB + kk + 16 + cseg, &Bs[buf][lbase + 16]);
    };

    v8f acc[2][4];
#pragma unroll
    for (int p = 0; p < 2; ++p)
#pragma unroll
        for (int i = 0; i < 4; ++i) acc[p][i] = (v8f){};

    // per-lane fragment offsets inside the LDS tiles
    const int aOff0 = (waveM * 32 + mr) * LDS_STRIDE + 8 * g;
    const int aOff1 = (waveM * 32 + 16 + mr) * LDS_STRIDE + 8 * g;

    auto compute = [&](int buf) {
        const unsigned short* as = &As[buf][0];
        const unsigned short* bs = &Bs[buf][0];
        FragBF a0, a1;
        a0.q[0] = *(const v4u*)(as + aOff0);
        a0.q[1] = *(const v4u*)(as + aOff0 + 16);
        a1.q[0] = *(const v4u*)(as + aOff1);
        a1.q[1] = *(const v4u*)(as + aOff1 + 16);
#pragma unroll
        for (int i = 0; i < 4; ++i) {
            const int bOff = (waveN * 64 + 16 * i + mr) * LDS_STRIDE + 8 * g;
            FragBF fb;
            fb.q[0] = *(const v4u*)(bs + bOff);
            fb.q[1] = *(const v4u*)(bs + bOff + 16);
            acc[0][i] = __builtin_amdgcn_wmma_f32_16x16x32_bf16(false, a0.v, false, fb.v,
                                                                (short)0, acc[0][i], false, false);
            acc[1][i] = __builtin_amdgcn_wmma_f32_16x16x32_bf16(false, a1.v, false, fb.v,
                                                                (short)0, acc[1][i], false, false);
        }
    };

    // double-buffered pipeline over 32 K-chunks
    stage(0, 0);
    stage(1, KCH);
    for (int c = 0; c < 31; ++c) {
        const int buf = c & 1;
#if USE_ASYNC_LDS
        __builtin_amdgcn_s_wait_asynccnt(4);   // chunk c retired; chunk c+1 in flight
#endif
        __syncthreads();
        compute(buf);
        __syncthreads();
        if (c < 30) stage(buf, (c + 2) * KCH);
    }
#if USE_ASYNC_LDS
    __builtin_amdgcn_s_wait_asynccnt(0);       // last chunk fully landed
#endif
    __syncthreads();
    compute(1);

    // --- epilogue: + b_out, scatter to logits ---
#pragma unroll
    for (int i = 0; i < 4; ++i) {
        const int n = nBase + 16 * i + mr;
        const float bo = b_out[n];
#pragma unroll
        for (int r = 0; r < 8; ++r) {
            logits[(long)(mBase + 8 * g + r) * VOCAB + n]      = acc[0][i][r] + bo;
            logits[(long)(mBase + 16 + 8 * g + r) * VOCAB + n] = acc[1][i][r] + bo;
        }
    }
}

// ---------------------------------------------------------------------------
// Cross-entropy per row: row_loss[r] = logsumexp(logits[r,:]) - logits[r, tgt]
// ---------------------------------------------------------------------------
__global__ __launch_bounds__(256) void ce_row(
    const float* __restrict__ logits, const int* __restrict__ targets,
    float* __restrict__ row_loss)
{
    const int r = blockIdx.x;
    const float* row = logits + (long)r * VOCAB;
    __shared__ float sred[256];

    float mx = -INFINITY;
    for (int i = threadIdx.x; i < VOCAB; i += 256) mx = fmaxf(mx, row[i]);
    sred[threadIdx.x] = mx;
    __syncthreads();
    for (int s = 128; s > 0; s >>= 1) {
        if (threadIdx.x < s) sred[threadIdx.x] = fmaxf(sred[threadIdx.x], sred[threadIdx.x + s]);
        __syncthreads();
    }
    mx = sred[0];
    __syncthreads();

    float sum = 0.0f;
    for (int i = threadIdx.x; i < VOCAB; i += 256) sum += expf(row[i] - mx);
    sred[threadIdx.x] = sum;
    __syncthreads();
    for (int s = 128; s > 0; s >>= 1) {
        if (threadIdx.x < s) sred[threadIdx.x] += sred[threadIdx.x + s];
        __syncthreads();
    }
    if (threadIdx.x == 0) {
        const float lse = logf(sred[0]) + mx;
        row_loss[r] = lse - row[targets[r]];   // targets flat [B,T] matches r = b*T+t
    }
}

__global__ __launch_bounds__(256) void loss_reduce(
    const float* __restrict__ row_loss, float* __restrict__ out)
{
    __shared__ float sred[256];
    float s = 0.0f;
    for (int i = threadIdx.x; i < ROWS; i += 256) s += row_loss[i];
    sred[threadIdx.x] = s;
    __syncthreads();
    for (int st = 128; st > 0; st >>= 1) {
        if (threadIdx.x < st) sred[threadIdx.x] += sred[threadIdx.x + st];
        __syncthreads();
    }
    if (threadIdx.x == 0) out[0] = sred[0] / (float)ROWS;
}

// ---------------------------------------------------------------------------
extern "C" void kernel_launch(void* const* d_in, const int* in_sizes, int n_in,
                              void* d_out, int out_size, void* d_ws, size_t ws_size,
                              hipStream_t stream)
{
    const int*   idx     = (const int*)  d_in[0];
    const int*   targets = (const int*)  d_in[1];
    const float* emb     = (const float*)d_in[2];
    const float* W_xh    = (const float*)d_in[3];
    const float* W_hh    = (const float*)d_in[4];
    const float* b_h     = (const float*)d_in[5];
    const float* W_out   = (const float*)d_in[6];
    const float* b_out   = (const float*)d_in[7];

    float* logits = (float*)d_out;                       // [B,T,V]
    float* loss   = (float*)d_out + (long)ROWS * VOCAB;  // final scalar

    // Workspace carve-out (256B aligned regions)
    uintptr_t base = (uintptr_t)d_ws;
    auto carve = [&](size_t bytes) {
        base = (base + 255) & ~(uintptr_t)255;
        uintptr_t p = base;
        base += bytes;
        return (void*)p;
    };
    unsigned short* WxhT  = (unsigned short*)carve((size_t)EMB * EMB * 2);    //  2 MB
    unsigned short* WoutT = (unsigned short*)carve((size_t)VOCAB * EMB * 2);  // 65 MB
    float*          xw    = (float*)carve((size_t)ROWS * EMB * 4);            // 16 MB
    unsigned short* hs    = (unsigned short*)carve((size_t)ROWS * EMB * 2);   //  8 MB
    float*          h_a   = (float*)carve((size_t)BB * EMB * 4);
    float*          h_b   = (float*)carve((size_t)BB * EMB * 4);
    float*          rls   = (float*)carve((size_t)ROWS * 4);

    // 1) Weight transpose/convert to bf16 [N][K]
    transpose_cvt<<<dim3(EMB / 32, EMB / 32), 256, 0, stream>>>(W_xh, WxhT, EMB, EMB);
    transpose_cvt<<<dim3(VOCAB / 32, EMB / 32), 256, 0, stream>>>(W_out, WoutT, EMB, VOCAB);

    // 2) h0 = 0
    zero_f32<<<(BB * EMB + 255) / 256, 256, 0, stream>>>(h_a, BB * EMB);

    // 3) Input projection (WMMA)
    input_proj<<<dim3(EMB / 64, ROWS / 64), 256, 0, stream>>>(idx, emb, WxhT, b_h, xw);

    // 4) Recurrence: 512 sequential steps (fp32), ping-pong h buffers
    float* hp = h_a;
    float* hn = h_b;
    for (int t = 0; t < TT; ++t) {
        rnn_step<<<(BB * EMB) / 256, 256, 0, stream>>>(xw, W_hh, hp, hn, hs, t);
        float* tmp = hp; hp = hn; hn = tmp;
    }

    // 5) Output projection (WMMA) -> logits
    out_gemm<<<dim3(VOCAB / 128, ROWS / 128), 256, 0, stream>>>(hs, WoutT, b_out, logits);

    // 6) Cross-entropy
    ce_row<<<ROWS, 256, 0, stream>>>(logits, targets, rls);
    loss_reduce<<<1, 256, 0, stream>>>(rls, loss);

    (void)in_sizes; (void)n_in; (void)out_size; (void)ws_size;
}
